// Encoder_54589034333003
// MI455X (gfx1250) — matrix operand order repaired and checked by reading, not compiled, not run on hardware
//
#include <hip/hip_runtime.h>
#include <cstdint>
#include <cstddef>

// ---------------- problem constants (from reference) ----------------
#define NB    4        // batch
#define NSEQ  16384    // E_TRI (T=1)
#define MR    (NB*NSEQ)// 65536 rows
#define DM    256      // model dim
#define NH    8
#define HDIM  32
#define DFF   1024
#define DQKV  768      // fused q|k|v output width

typedef __attribute__((ext_vector_type(16))) __bf16 v16bf;
typedef __attribute__((ext_vector_type(8)))  float  v8f;

union Frag32 { v16bf v; uint4 u[2]; };

__device__ __forceinline__ unsigned short f32_to_bf16(float f) {
    unsigned int u = __float_as_uint(f);
    u += 0x7FFFu + ((u >> 16) & 1u);          // round to nearest even
    return (unsigned short)(u >> 16);
}

// ---------------- weight convert + transpose: W[fi][fo] -> WT[fo][fi] bf16 ----
__global__ void convT_kernel(const float* __restrict__ W, unsigned short* __restrict__ WT,
                             int fi, int fo) {
    int t = blockIdx.x * 256 + threadIdx.x;
    if (t >= fi * fo) return;
    int kk = t % fi;
    int nn = t / fi;
    WT[(size_t)nn * fi + kk] = f32_to_bf16(W[(size_t)kk * fo + nn]);
}

// ---------------- small f32 copy (bias packing) ----------------
__global__ void copyf_kernel(const float* __restrict__ src, float* __restrict__ dst, int n) {
    int i = blockIdx.x * 256 + threadIdx.x;
    if (i < n) dst[i] = src[i];
}

// ---------------- embed: h = x @ We + be + pos ----------------
__global__ __launch_bounds__(256) void embed_kernel(const float* __restrict__ x,
                                                    const float* __restrict__ we,
                                                    const float* __restrict__ be,
                                                    const float* __restrict__ pos,
                                                    float* __restrict__ h) {
    size_t t = (size_t)blockIdx.x * 256 + threadIdx.x;   // MR*DM threads
    int    d = (int)(t & (DM - 1));
    size_t m = t >> 8;
    int    n = (int)(m & (NSEQ - 1));
    float x0 = x[m * 2 + 0];
    float x1 = x[m * 2 + 1];
    h[t] = x0 * we[d] + x1 * we[DM + d] + be[d] + pos[(size_t)n * DM + d];
}

// ---------------- LayerNorm: wave per row of 256; bf16 or f32 output ----------
__global__ __launch_bounds__(256) void ln_kernel(const float* __restrict__ x,
                                                 const float* __restrict__ w,
                                                 const float* __restrict__ b,
                                                 unsigned short* __restrict__ outB,
                                                 float* __restrict__ outF,
                                                 int Mrows) {
    int wid  = (int)((blockIdx.x * blockDim.x + threadIdx.x) >> 5);
    int lane = threadIdx.x & 31;
    if (wid >= Mrows) return;
    const float* row = x + (size_t)wid * DM;
    float vals[8], s = 0.f, sq = 0.f;
#pragma unroll
    for (int i = 0; i < 8; i++) {
        float f = row[lane * 8 + i];
        vals[i] = f; s += f; sq += f * f;
    }
#pragma unroll
    for (int off = 16; off >= 1; off >>= 1) {
        s  += __shfl_xor(s,  off, 32);
        sq += __shfl_xor(sq, off, 32);
    }
    float mean = s * (1.f / DM);
    float var  = sq * (1.f / DM) - mean * mean;
    float inv  = rsqrtf(var + 1e-5f);
#pragma unroll
    for (int i = 0; i < 8; i++) {
        int d = lane * 8 + i;
        float o = (vals[i] - mean) * inv * w[d] + b[d];
        if (outB) outB[(size_t)wid * DM + d] = f32_to_bf16(o);
        else      outF[(size_t)wid * DM + d] = o;
    }
}

// ---------------- bf16 WMMA GEMM: C[M][Nout] = A[M][K] * WT[Nout][K]^T -------
// wave = 32 rows x 64 cols (2 M-tiles x 4 N-tiles = 8 accumulators);
// B fragments reused across both M-tiles -> 1.5 b128 loads per WMMA.
// fused bias / exact-GELU / residual epilogue; f32 or bf16 output.
template <int K>
__global__ __launch_bounds__(256) void gemm_kernel(const unsigned short* __restrict__ A,
                                                   const unsigned short* __restrict__ WT,
                                                   const float* __restrict__ bias,
                                                   const float* __restrict__ resid,
                                                   float* __restrict__ outF,
                                                   unsigned short* __restrict__ outB,
                                                   int Nout, int gelu) {
    const int warp = threadIdx.x >> 5;
    const int lane = threadIdx.x & 31;
    const int hl   = lane >> 4;      // half-wave selector
    const int l16  = lane & 15;
    const int ngroups = Nout >> 6;   // 64-col groups
    int gw    = blockIdx.x * 8 + warp;
    int mpair = gw / ngroups;
    int ngrp  = gw - mpair * ngroups;
    int m0    = mpair << 5;          // 32 rows per wave
    int n0    = ngrp << 6;

    const unsigned short* Arow0 = A + (size_t)(m0 + l16) * K;
    const unsigned short* Arow1 = Arow0 + (size_t)16 * K;

    v8f c0[4] = {};
    v8f c1[4] = {};

#pragma unroll 2
    for (int kb = 0; kb < K; kb += 32) {
        // A fragments (ISA 16-bit A layout): lanes 0-15 K=kb..+7 & kb+16..+23,
        // lanes 16-31 shifted by 8.
        Frag32 a0, a1;
        {
            const unsigned short* ap0 = Arow0 + kb + hl * 8;
            a0.u[0] = *(const uint4*)(ap0);
            a0.u[1] = *(const uint4*)(ap0 + 16);
            const unsigned short* ap1 = Arow1 + kb + hl * 8;
            a1.u[0] = *(const uint4*)(ap1);
            a1.u[1] = *(const uint4*)(ap1 + 16);
        }
#pragma unroll
        for (int t = 0; t < 4; t++) {
            // B fragment: col n0+16t+l16; lanes 0-15 K=kb..+7 & kb+16..+23, lanes 16-31 shifted by 8
            Frag32 bf;
            const unsigned short* bp = WT + (size_t)(n0 + t * 16 + l16) * K + kb + hl * 8;
            bf.u[0] = *(const uint4*)(bp);
            bf.u[1] = *(const uint4*)(bp + 16);
            c0[t] = __builtin_amdgcn_wmma_f32_16x16x32_bf16(
                false, a0.v, false, bf.v, (short)0, c0[t], false, false);
            c1[t] = __builtin_amdgcn_wmma_f32_16x16x32_bf16(
                false, a1.v, false, bf.v, (short)0, c1[t], false, false);
        }
    }

    // epilogue: C layout — VGPR g holds row base + g + hl*8, col n0+16t+l16
#pragma unroll
    for (int t = 0; t < 4; t++) {
        int   col  = n0 + t * 16 + l16;
        float bcol = bias[col];
#pragma unroll
        for (int g = 0; g < 8; g++) {
#pragma unroll
            for (int mt = 0; mt < 2; mt++) {
                int   rowm = m0 + mt * 16 + hl * 8 + g;
                float val  = (mt ? c1[t][g] : c0[t][g]) + bcol;
                if (gelu)  val = 0.5f * val * (1.f + erff(val * 0.70710678118f));
                if (resid) val += resid[(size_t)rowm * Nout + col];
                if (outF)  outF[(size_t)rowm * Nout + col] = val;
                else       outB[(size_t)rowm * Nout + col] = f32_to_bf16(val);
            }
        }
    }
}

// ---------------- neighbor attention on fused QKV (M x 768) ------------------
// wave per (b,n), lane = head-dim; q at +0, k at +256, v at +512 within a row.
__global__ __launch_bounds__(256) void attn_kernel(const float* __restrict__ qkv,
                                                   const int* __restrict__ nt,
                                                   unsigned short* __restrict__ outB) {
    int gw   = (int)((blockIdx.x * blockDim.x + threadIdx.x) >> 5);   // b*N+n
    int lane = threadIdx.x & 31;
    int bi   = gw >> 14;
    int n    = gw & (NSEQ - 1);
    size_t qrow  = (size_t)gw * DQKV;
    size_t bbase = (size_t)bi * NSEQ;

    int j0 = nt[n], j1 = nt[NSEQ + n], j2 = nt[2 * NSEQ + n];
    int v0 = j0 >= 0, v1 = j1 >= 0, v2 = j2 >= 0;
    size_t r0 = (bbase + (size_t)max(j0, 0)) * DQKV;
    size_t r1 = (bbase + (size_t)max(j1, 0)) * DQKV;
    size_t r2 = (bbase + (size_t)max(j2, 0)) * DQKV;

    const float scale = 0.17677669529663687f; // 1/sqrt(32)
    for (int h = 0; h < NH; h++) {
        int off = h * HDIM + lane;
        float qh = qkv[qrow + off];
        float k0 = qkv[r0 + DM + off], k1 = qkv[r1 + DM + off], k2 = qkv[r2 + DM + off];
        float p0 = qh * k0, p1 = qh * k1, p2 = qh * k2;
#pragma unroll
        for (int o = 16; o >= 1; o >>= 1) {
            p0 += __shfl_xor(p0, o, 32);
            p1 += __shfl_xor(p1, o, 32);
            p2 += __shfl_xor(p2, o, 32);
        }
        float s0 = v0 ? p0 * scale : -1e30f;
        float s1 = v1 ? p1 * scale : -1e30f;
        float s2 = v2 ? p2 * scale : -1e30f;
        float mx = fmaxf(s0, fmaxf(s1, s2));
        float e0 = v0 ? expf(s0 - mx) : 0.f;
        float e1 = v1 ? expf(s1 - mx) : 0.f;
        float e2 = v2 ? expf(s2 - mx) : 0.f;
        float inv = 1.f / (e0 + e1 + e2);
        float o = (e0 * qkv[r0 + 2 * DM + off] + e1 * qkv[r1 + 2 * DM + off] +
                   e2 * qkv[r2 + 2 * DM + off]) * inv;
        outB[(size_t)gw * DM + off] = f32_to_bf16(o);
    }
}

// ---------------- host orchestration ----------------
extern "C" void kernel_launch(void* const* d_in, const int* in_sizes, int n_in,
                              void* d_out, int out_size, void* d_ws, size_t ws_size,
                              hipStream_t stream) {
    (void)in_sizes; (void)n_in; (void)out_size; (void)ws_size;

    const float* x    = (const float*)d_in[0];
    const int*   nt   = (const int*)d_in[1];     // jax default x64-off -> int32
    const float* ew   = (const float*)d_in[2];
    const float* eb   = (const float*)d_in[3];
    const float* pos  = (const float*)d_in[4];
    const float* lnw  = (const float*)d_in[37];
    const float* lnb  = (const float*)d_in[38];

    // ---- workspace layout ----
    char* base = (char*)d_ws;
    size_t off = 0;
    auto alloc = [&](size_t bytes) -> char* {
        char* p = base + off;
        off = (off + bytes + 255) & ~(size_t)255;
        return p;
    };
    unsigned short* WTqkv[2]; unsigned short* WTo[2];
    unsigned short* WTm1[2];  unsigned short* WTm2[2];
    float* B768[2];
    for (int L = 0; L < 2; L++) {
        WTqkv[L] = (unsigned short*)alloc((size_t)3 * DM * DM * 2); // q|k|v rows packed
        WTo[L]   = (unsigned short*)alloc((size_t)DM * DM * 2);
        WTm1[L]  = (unsigned short*)alloc((size_t)DM * DFF * 2);
        WTm2[L]  = (unsigned short*)alloc((size_t)DFF * DM * 2);
        B768[L]  = (float*)alloc((size_t)DQKV * 4);
    }
    unsigned short* XH = (unsigned short*)alloc((size_t)MR * DM * 2);   // bf16 GEMM-A staging
    float*          H  = (float*)alloc((size_t)MR * DM * 4);            // residual stream
    char* U = alloc((size_t)MR * DQKV * 4);                             // union region (192MB)
    float*          QKV = (float*)U;                                    // attention phase (M x 768 f32)
    unsigned short* MID = (unsigned short*)U;                           // MLP phase (M x 1024 bf16)

    // ---- convert / transpose weights + pack qkv biases (tiny) ----
    for (int L = 0; L < 2; L++) {
        int pb = 5 + L * 16;
        convT_kernel<<<(DM * DM + 255) / 256, 256, 0, stream>>>((const float*)d_in[pb + 2],  WTqkv[L],               DM, DM);
        convT_kernel<<<(DM * DM + 255) / 256, 256, 0, stream>>>((const float*)d_in[pb + 4],  WTqkv[L] + DM * DM,     DM, DM);
        convT_kernel<<<(DM * DM + 255) / 256, 256, 0, stream>>>((const float*)d_in[pb + 6],  WTqkv[L] + 2 * DM * DM, DM, DM);
        convT_kernel<<<(DM * DM + 255) / 256, 256, 0, stream>>>((const float*)d_in[pb + 8],  WTo[L],  DM, DM);
        convT_kernel<<<(DM * DFF + 255) / 256, 256, 0, stream>>>((const float*)d_in[pb + 12], WTm1[L], DM,  DFF);
        convT_kernel<<<(DFF * DM + 255) / 256, 256, 0, stream>>>((const float*)d_in[pb + 14], WTm2[L], DFF, DM);
        copyf_kernel<<<1, 256, 0, stream>>>((const float*)d_in[pb + 3], B768[L],          DM);
        copyf_kernel<<<1, 256, 0, stream>>>((const float*)d_in[pb + 5], B768[L] + DM,     DM);
        copyf_kernel<<<1, 256, 0, stream>>>((const float*)d_in[pb + 7], B768[L] + 2 * DM, DM);
    }

    // ---- embed + pos ----
    embed_kernel<<<(MR * DM) / 256, 256, 0, stream>>>(x, ew, eb, pos, H);

    const int LN_BLOCKS   = MR / 8;                           // wave per row
    const int AT_BLOCKS   = MR / 8;
    const int GQKV_BLOCKS = (MR / 32) * (DQKV / 64) / 8;      // 3072
    const int G256_BLOCKS = (MR / 32) * (DM   / 64) / 8;      // 1024
    const int G1K_BLOCKS  = (MR / 32) * (DFF  / 64) / 8;      // 4096

    for (int L = 0; L < 2; L++) {
        int pb = 5 + L * 16;
        const float* ln1w = (const float*)d_in[pb + 0];
        const float* ln1b = (const float*)d_in[pb + 1];
        const float* ob   = (const float*)d_in[pb + 9];
        const float* ln2w = (const float*)d_in[pb + 10];
        const float* ln2b = (const float*)d_in[pb + 11];
        const float* m1b  = (const float*)d_in[pb + 13];
        const float* m2b  = (const float*)d_in[pb + 15];

        // attention block (fused QKV projection)
        ln_kernel<<<LN_BLOCKS, 256, 0, stream>>>(H, ln1w, ln1b, XH, nullptr, MR);
        gemm_kernel<DM><<<GQKV_BLOCKS, 256, 0, stream>>>(XH, WTqkv[L], B768[L], nullptr, QKV, nullptr, DQKV, 0);
        attn_kernel<<<AT_BLOCKS, 256, 0, stream>>>(QKV, nt, XH);
        gemm_kernel<DM><<<G256_BLOCKS, 256, 0, stream>>>(XH, WTo[L], ob, H, H, nullptr, DM, 0);

        // MLP block
        ln_kernel<<<LN_BLOCKS, 256, 0, stream>>>(H, ln2w, ln2b, XH, nullptr, MR);
        gemm_kernel<DM><<<G1K_BLOCKS, 256, 0, stream>>>(XH, WTm1[L], m1b, nullptr, nullptr, MID, DFF, 1);
        gemm_kernel<DFF><<<G256_BLOCKS, 256, 0, stream>>>(MID, WTm2[L], m2b, H, H, nullptr, DM, 0);
    }

    // final LayerNorm -> d_out (fp32)
    ln_kernel<<<LN_BLOCKS, 256, 0, stream>>>(H, lnw, lnb, nullptr, (float*)d_out, MR);
}
